// PatchScanMamba_52278341927352
// MI455X (gfx1250) — matrix operand order repaired and checked by reading, not compiled, not run on hardware
//
#include <hip/hip_runtime.h>
#include <hip/hip_bf16.h>
#include <cstdint>
#include <cstddef>

#define NPATCH 192
#define BATCH  32
#define SEQQ   193
#define MROWS  (BATCH*NPATCH)   /* 6144 */

typedef __attribute__((ext_vector_type(16))) _Float16 v16h;
typedef __attribute__((ext_vector_type(8)))  _Float16 v8h;
typedef __attribute__((ext_vector_type(8)))  float    v8f;

__device__ __forceinline__ float silu_f(float x){ return x * (1.f/(1.f + __expf(-x))); }
__device__ __forceinline__ float softplus_f(float x){ return x > 20.f ? x : log1pf(__expf(x)); }

// ---------------- fp32 -> fp16 weight conversion ----------------
__global__ __launch_bounds__(256) void f2h_kernel(const float* __restrict__ s,
                                                  _Float16* __restrict__ d, int n){
  int i = blockIdx.x*256 + threadIdx.x;
  if(i < n) d[i] = (_Float16)s[i];
}

// ---------------- gather(spiral) + LayerNorm (both directions) ----------------
__global__ __launch_bounds__(128) void ln_gather_kernel(
    const float* __restrict__ x, const int* __restrict__ sidx,
    const float* __restrict__ gf, const float* __restrict__ bf,
    const float* __restrict__ gb, const float* __restrict__ bb,
    float* __restrict__ pr, _Float16* __restrict__ xlnF, _Float16* __restrict__ xlnB){
  int m = blockIdx.x;            // 0..6143
  int b = m / NPATCH, l = m % NPATCH;
  int tid = threadIdx.x;
  __shared__ float v[384];
  __shared__ float red[128];
  __shared__ float stat[2];
  const float* src = x + ((size_t)b*SEQQ + 1 + sidx[l]) * 384;
  float* prr = pr + (size_t)m*384;
  for(int i=tid;i<384;i+=128){ float t = src[i]; v[i]=t; prr[i]=t; }
  __syncthreads();
  float s = 0.f;
  for(int i=tid;i<384;i+=128) s += v[i];
  red[tid]=s; __syncthreads();
  for(int o=64;o>0;o>>=1){ if(tid<o) red[tid]+=red[tid+o]; __syncthreads(); }
  if(tid==0) stat[0] = red[0]*(1.f/384.f);
  __syncthreads();
  float mean = stat[0];
  s = 0.f;
  for(int i=tid;i<384;i+=128){ float t=v[i]-mean; s += t*t; }
  red[tid]=s; __syncthreads();
  for(int o=64;o>0;o>>=1){ if(tid<o) red[tid]+=red[tid+o]; __syncthreads(); }
  if(tid==0) stat[1] = red[0]*(1.f/384.f);
  __syncthreads();
  float rstd = rsqrtf(stat[1] + 1e-5f);
  _Float16* xfr = xlnF + (size_t)m*384;
  _Float16* xbr = xlnB + ((size_t)(b*NPATCH + (NPATCH-1-l)))*384;  // flipped seq
  for(int i=tid;i<384;i+=128){
    float xn = (v[i]-mean)*rstd;
    xfr[i] = (_Float16)(xn*gf[i] + bf[i]);
    xbr[i] = (_Float16)(xn*gb[i] + bb[i]);
  }
}

// ---------------- register-blocked WMMA f16 GEMM: C[M,N] = A[M,K] * W[N,K]^T ----
// Each wave computes a 32x64 tile: 2 M-subtiles x 4 N-subtiles (8x v8f acc).
// Per 32-K step: 4 A-loads + 8 B-loads feed 8 wmma (1.5 b128 loads / wmma).
// mode 0: store f32 to outF (ld=ldOut)
// mode 1: acc + resid[orow]  -> f16 into outH[orow*768 + colOff + n]  (orow = unflipped)
// mode 2: acc + bias[n]      -> f32 scatter to dOut[b,1+sidx[l],n]
__global__ __launch_bounds__(256) void gemm16_kernel(
    const _Float16* __restrict__ A, const _Float16* __restrict__ W,
    int M, int N, int K, int mode,
    float* __restrict__ outF, int ldOut,
    _Float16* __restrict__ outH, const float* __restrict__ resid, int colOff, int flip,
    const float* __restrict__ bias, const int* __restrict__ sidx, float* __restrict__ dOut){
  int wave = (blockIdx.x*256 + threadIdx.x) >> 5;
  int lane = threadIdx.x & 31;
  int tilesN64 = N >> 6;
  int tM = wave / tilesN64;
  int tN = wave - tM*tilesN64;
  if(tM >= (M >> 5)) return;           // wave-uniform
  int m0 = tM << 5, n0 = tN << 6;
  int lh = lane & 15, ls = lane >> 4;
  const _Float16* ar = A + (size_t)(m0+lh)*K;   // +mt*16*K per M-subtile
  const _Float16* wr = W + (size_t)(n0+lh)*K;   // +nt*16*K per N-subtile
  v8f acc[2][4];
#pragma unroll
  for(int mt=0;mt<2;mt++)
#pragma unroll
    for(int nt=0;nt<4;nt++)
#pragma unroll
      for(int i=0;i<8;i++) acc[mt][nt][i]=0.f;

  for(int k0=0;k0<K;k0+=32){
    v16h af[2]; v16h bf[4];
#pragma unroll
    for(int mt=0;mt<2;mt++){
      // A 16x32 fragment: lane<16 -> K[k0..k0+15]; lane>=16 -> K[k0+16..k0+31] (contiguous)
      const _Float16* arm = ar + (size_t)mt*16*K;
      af[mt] = *(const v16h*)(arm + k0 + ls*16);
    }
#pragma unroll
    for(int nt=0;nt<4;nt++){
      // B 32x16 fragment: lane n holds W[n0+nt*16+n][k0 + 16*ls .. +15] (contiguous)
      bf[nt] = *(const v16h*)(wr + (size_t)nt*16*K + k0 + ls*16);
    }
#pragma unroll
    for(int mt=0;mt<2;mt++)
#pragma unroll
      for(int nt=0;nt<4;nt++)
        acc[mt][nt] = __builtin_amdgcn_wmma_f32_16x16x32_f16(
            false, af[mt], false, bf[nt], (short)0, acc[mt][nt], false, false);
  }

  int c = lh;  // output column within 16-wide subtile
#pragma unroll
  for(int mt=0;mt<2;mt++){
#pragma unroll
    for(int nt=0;nt<4;nt++){
      int ms = m0 + mt*16;
      int ns = n0 + nt*16;
      if(mode == 0){
#pragma unroll
        for(int r=0;r<8;r++){
          int mr = ms + r + 8*ls;
          outF[(size_t)mr*ldOut + ns + c] = acc[mt][nt][r];
        }
      } else if(mode == 1){
#pragma unroll
        for(int r=0;r<8;r++){
          int mr = ms + r + 8*ls;
          int b = mr / NPATCH, sgm = mr % NPATCH;
          int orow = flip ? (b*NPATCH + (NPATCH-1-sgm)) : mr;
          float val = acc[mt][nt][r] + resid[(size_t)orow*384 + ns + c];
          outH[(size_t)orow*768 + colOff + ns + c] = (_Float16)val;
        }
      } else {
#pragma unroll
        for(int r=0;r<8;r++){
          int mr = ms + r + 8*ls;
          int b = mr / NPATCH, l = mr % NPATCH;
          int dest = b*SEQQ + 1 + sidx[l];
          dOut[(size_t)dest*384 + ns + c] = acc[mt][nt][r] + bias[ns + c];
        }
      }
    }
  }
}

// ---------------- causal depthwise conv (k=4) + SiLU ----------------
__global__ __launch_bounds__(256) void conv_silu_kernel(
    const float* __restrict__ xz, const float* __restrict__ cw,
    const float* __restrict__ cb, float* __restrict__ xc){
  int i = blockIdx.x*256 + threadIdx.x;
  if(i >= MROWS*768) return;
  int d = i % 768;
  int m = i / 768;
  int l = m % NPATCH, b = m / NPATCH;
  float a = cb[d];
#pragma unroll
  for(int j=0;j<4;j++){
    int sl = l - 3 + j;
    if(sl >= 0) a += cw[d*4+j] * xz[(size_t)(b*NPATCH+sl)*1536 + d];
  }
  xc[(size_t)m*768 + d] = silu_f(a);
}

// ---------------- x-proj (56 outs) + dt-proj(+softplus) per row ----------------
__global__ __launch_bounds__(256) void xproj_dt_kernel(
    const float* __restrict__ xc, const float* __restrict__ x_w,
    const float* __restrict__ dt_w, const float* __restrict__ dt_b,
    float* __restrict__ xdbl, float* __restrict__ dt){
  int m = blockIdx.x;
  int tid = threadIdx.x;
  __shared__ float u[768];
  __shared__ float dtr[24];
  for(int i=tid;i<768;i+=256) u[i] = xc[(size_t)m*768 + i];
  __syncthreads();
  if(tid < 56){
    float a = 0.f;
    const float* wr = x_w + (size_t)tid*768;
    for(int d=0;d<768;d++) a += u[d]*wr[d];
    xdbl[(size_t)m*56 + tid] = a;
    if(tid < 24) dtr[tid] = a;
  }
  __syncthreads();
  for(int d=tid; d<768; d+=256){
    float a = dt_b[d];
    const float* wr = dt_w + (size_t)d*24;
#pragma unroll
    for(int r=0;r<24;r++) a += dtr[r]*wr[r];
    dt[(size_t)m*768 + d] = softplus_f(a);
  }
}

// ------- selective scan (sequential over L) + fused SiLU(z) gate, emits f16 -------
__global__ __launch_bounds__(256) void scan_gate_kernel(
    const float* __restrict__ u, const float* __restrict__ dt,
    const float* __restrict__ xdbl, const float* __restrict__ A_log,
    const float* __restrict__ Dp, const float* __restrict__ xz,
    _Float16* __restrict__ yg){
  int d = blockIdx.x*256 + threadIdx.x;   // gridDim.x = 3 -> d < 768
  int b = blockIdx.y;
  float Ac[16], h[16];
#pragma unroll
  for(int n=0;n<16;n++){ Ac[n] = -__expf(A_log[(size_t)d*16+n]); h[n]=0.f; }
  float Dd = Dp[d];
  for(int l=0;l<NPATCH;l++){
    size_t row = (size_t)b*NPATCH + l;
    float dtv = dt[row*768 + d];
    float uv  = u [row*768 + d];
    const float* Bm = xdbl + row*56 + 24;
    const float* Cm = Bm + 16;
    float dbu = dtv*uv;
    float acc = 0.f;
#pragma unroll
    for(int n=0;n<16;n++){
      h[n] = __expf(dtv*Ac[n])*h[n] + dbu*Bm[n];
      acc += h[n]*Cm[n];
    }
    float y = acc + uv*Dd;
    float z = xz[row*1536 + 768 + d];
    yg[row*768 + d] = (_Float16)(y * silu_f(z));
  }
}

// ---------------- cls token passthrough ----------------
__global__ __launch_bounds__(128) void cls_kernel(const float* __restrict__ x,
                                                  float* __restrict__ out){
  int b = blockIdx.x;
  for(int i=threadIdx.x;i<384;i+=128)
    out[(size_t)b*SEQQ*384 + i] = x[(size_t)b*SEQQ*384 + i];
}

extern "C" void kernel_launch(void* const* d_in, const int* in_sizes, int n_in,
                              void* d_out, int out_size, void* d_ws, size_t ws_size,
                              hipStream_t stream){
  (void)in_sizes; (void)n_in; (void)out_size; (void)ws_size;
  const float* x = (const float*)d_in[0];
  // per-direction params: f_* at 1..11, b_* at 12..22
  const float* ln_g[2]  = {(const float*)d_in[1],  (const float*)d_in[12]};
  const float* ln_b[2]  = {(const float*)d_in[2],  (const float*)d_in[13]};
  const float* in_w[2]  = {(const float*)d_in[3],  (const float*)d_in[14]};
  const float* conv_w[2]= {(const float*)d_in[4],  (const float*)d_in[15]};
  const float* conv_b[2]= {(const float*)d_in[5],  (const float*)d_in[16]};
  const float* x_w[2]   = {(const float*)d_in[6],  (const float*)d_in[17]};
  const float* dt_w[2]  = {(const float*)d_in[7],  (const float*)d_in[18]};
  const float* dt_b[2]  = {(const float*)d_in[8],  (const float*)d_in[19]};
  const float* A_log[2] = {(const float*)d_in[9],  (const float*)d_in[20]};
  const float* Dp[2]    = {(const float*)d_in[10], (const float*)d_in[21]};
  const float* out_w[2] = {(const float*)d_in[11], (const float*)d_in[22]};
  const float* fusion_w = (const float*)d_in[23];
  const float* fusion_b = (const float*)d_in[24];
  const int*   scan_idx = (const int*)d_in[25];
  float* out = (float*)d_out;

  // ---- workspace carving (all 256-B aligned) ----
  char* w = (char*)d_ws;
  size_t off = 0;
  auto carve = [&](size_t bytes)->char*{
    char* p = w + off;
    off = (off + bytes + 255) & ~(size_t)255;
    return p;
  };
  float*    pr     = (float*)   carve((size_t)MROWS*384*4);
  _Float16* xlnF   = (_Float16*)carve((size_t)MROWS*384*2);
  _Float16* xlnB   = (_Float16*)carve((size_t)MROWS*384*2);
  _Float16* inw16f = (_Float16*)carve((size_t)1536*384*2);
  _Float16* inw16b = (_Float16*)carve((size_t)1536*384*2);
  _Float16* outw16f= (_Float16*)carve((size_t)384*768*2);
  _Float16* outw16b= (_Float16*)carve((size_t)384*768*2);
  _Float16* fusw16 = (_Float16*)carve((size_t)384*768*2);
  float*    xz     = (float*)   carve((size_t)MROWS*1536*4);   // reused per dir
  float*    xc     = (float*)   carve((size_t)MROWS*768*4);    // conv out / scan u
  float*    xdbl   = (float*)   carve((size_t)MROWS*56*4);     // dtr|B|C
  float*    dt     = (float*)   carve((size_t)MROWS*768*4);    // dt (softplus)
  _Float16* yg     = (_Float16*)carve((size_t)MROWS*768*2);    // gated y (f16)
  _Float16* fused  = (_Float16*)carve((size_t)MROWS*768*2);    // concat(out_f,out_b)

  _Float16* inw16[2]  = {inw16f, inw16b};
  _Float16* outw16[2] = {outw16f, outw16b};
  _Float16* xln[2]    = {xlnF, xlnB};

  // ---- weight conversion fp32 -> fp16 ----
  for(int d=0; d<2; ++d){
    f2h_kernel<<<(1536*384+255)/256,256,0,stream>>>(in_w[d],  inw16[d],  1536*384);
    f2h_kernel<<<(384*768+255)/256, 256,0,stream>>>(out_w[d], outw16[d], 384*768);
  }
  f2h_kernel<<<(384*768+255)/256,256,0,stream>>>(fusion_w, fusw16, 384*768);

  // ---- gather + LN (both directions at once) ----
  ln_gather_kernel<<<MROWS,128,0,stream>>>(x, scan_idx,
      ln_g[0], ln_b[0], ln_g[1], ln_b[1], pr, xlnF, xlnB);

  const int ew_blocks = (MROWS*768)/256;                 // 18432
  const int g1_blocks = ((MROWS/32)*(1536/64))/8;        // 576 (8 waves/block)
  const int g2_blocks = ((MROWS/32)*(384/64))/8;         // 144

  for(int dir=0; dir<2; ++dir){
    // in-projection: xz = xln @ in_w^T   [6144 x 1536]
    gemm16_kernel<<<g1_blocks,256,0,stream>>>(xln[dir], inw16[dir],
        MROWS, 1536, 384, /*mode*/0, xz, 1536,
        nullptr, nullptr, 0, 0, nullptr, nullptr, nullptr);
    // depthwise causal conv + SiLU on xm half
    conv_silu_kernel<<<ew_blocks,256,0,stream>>>(xz, conv_w[dir], conv_b[dir], xc);
    // x-proj (dtr|B|C) + dt-proj + softplus
    xproj_dt_kernel<<<MROWS,256,0,stream>>>(xc, x_w[dir], dt_w[dir], dt_b[dir],
                                            xdbl, dt);
    // selective scan + fused SiLU(z) gate -> f16
    scan_gate_kernel<<<dim3(3,BATCH),256,0,stream>>>(xc, dt, xdbl,
        A_log[dir], Dp[dir], xz, yg);
    // out-projection + residual, un-flip for backward, pack into fusion buffer
    gemm16_kernel<<<g2_blocks,256,0,stream>>>(yg, outw16[dir],
        MROWS, 384, 768, /*mode*/1, nullptr, 0,
        fused, pr, /*colOff*/dir*384, /*flip*/dir, nullptr, nullptr, nullptr);
  }

  // fusion GEMM + bias, scatter rows to un-spiral order in d_out
  gemm16_kernel<<<g2_blocks,256,0,stream>>>(fused, fusw16,
      MROWS, 384, 768, /*mode*/2, nullptr, 0,
      nullptr, nullptr, 0, 0, fusion_b, scan_idx, out);

  // cls token passthrough
  cls_kernel<<<BATCH,128,0,stream>>>(x, out);
}
